// TemPred_74672301408343
// MI455X (gfx1250) — matrix operand order repaired and checked by reading, not compiled, not run on hardware
//
#include <hip/hip_runtime.h>

// Problem dims (fixed by reference)
#define BDIM  2048
#define TTOT  168
#define TEMPD 128
#define HID   1024
#define G4H   4096   // 4*H
#define MDEC  48

// GEMM tiling
#define TM 128
#define TN 128
#define TK 32
#define LSTR 40      // LDS row stride in bf16 elements (80B, keeps 16B alignment)
#define BUFSZ (TM * LSTR)   // one LDS buffer in elements

typedef __attribute__((ext_vector_type(16))) __bf16 v16bf;
typedef __attribute__((ext_vector_type(8)))  float  v8f;

__device__ __forceinline__ unsigned short f2bf(float f) {
    unsigned int u = __float_as_uint(f);
    unsigned int r = u + 0x7FFFu + ((u >> 16) & 1u);   // round-to-nearest-even
    return (unsigned short)(r >> 16);
}

// Gate-interleaved permutation of the 4H output dim:
//   permuted row nr  <-  source row g*H + j,  where nr = (j/16)*64 + g*16 + (j%16)
__device__ __forceinline__ int perm_src_row(int nr) {
    int jgrp = nr >> 6, rem = nr & 63;
    int g = rem >> 4, jl = rem & 15;
    return g * HID + jgrp * 16 + jl;
}

// Low 32 bits of a flat LDS address are the LDS byte offset (ISA 10.2).
__device__ __forceinline__ unsigned lds_off(const void* p) {
    return (unsigned)(unsigned long long)p;
}

// CDNA5 async global->LDS copy: each lane moves 16B; tracked by ASYNCcnt.
__device__ __forceinline__ void async_b128(unsigned lds, const void* gaddr) {
    asm volatile("global_load_async_to_lds_b128 %0, %1, off"
                 :: "v"(lds), "v"(gaddr)
                 : "memory");
}
__device__ __forceinline__ void async_wait0() {
    asm volatile("s_wait_asynccnt 0" ::: "memory");
}

// ---------------------------------------------------------------------------
// Double-buffered, async-staged GEMM mainloop over two K-segments:
//   acc += X @ Wx^T   (KX wide; f32-convert or bf16-async path)
//   acc += H @ Wh^T   (KH wide; bf16 async path)
// One barrier per K-step: stage(kk+1) is issued async while computing kk.
// Buffer selection is by INDEX OFFSET into single doubled __shared__ arrays
// (never by pointer select) so addrspace(3) inference keeps ds_load/ds_store.
// 8 wave32s: wm = wid>>1 (32-row slab), wn = wid&1 (64-col slab).
// acc[mi*4+g]: mi in {0,1}, g in {0..3} -> v_wmma_f32_16x16x32_bf16 x8 / step.
// ---------------------------------------------------------------------------
template <typename XT>
__device__ __forceinline__ void gemm_pipelined(
    const XT* __restrict__ X, int ldx,
    const unsigned short* __restrict__ Wx, int ldbx, int KX,
    const unsigned short* __restrict__ Hm, int ldh,
    const unsigned short* __restrict__ Wh, int ldbh, int KH,
    int mbase, int nbase, int tid,
    unsigned short* AsD,      // [2 * BUFSZ] shared
    unsigned short* BsD,      // [2 * BUFSZ] shared
    v8f* acc)
{
    const int lane = tid & 31;
    const int wid  = tid >> 5;
    const int wm   = wid >> 1;           // 0..3
    const int wn   = wid & 1;            // 0..1
    const int lrow = tid >> 1;           // 0..127 (staging row)
    const int lcol = (tid & 1) * 16;     // 0 or 16

    const int xsteps = KX / TK;
    const int NS     = xsteps + KH / TK;

    // CDNA5 ISA 7.12.2 fragment layouts
    const int aoff = (lane < 16) ? 0 : 8;   // A: lanes16-31 hold K 8-15 / 24-31
    const int arow = lane & 15;
    const int boff = (lane < 16) ? 0 : 8;   // B: lanes16-31 hold K 8-15 / 24-31
    const int bcol = lane & 15;

    auto stage = [&](int kk) {
        const int soff = (kk & 1) * BUFSZ;              // index offset, not pointer select
        const unsigned aLds = lds_off(&AsD[soff + lrow * LSTR + lcol]);
        const unsigned bLds = lds_off(&BsD[soff + lrow * LSTR + lcol]);
        if (kk < xsteps) {
            const int k0 = kk * TK;
            if constexpr (sizeof(XT) == 4) {
                // f32 -> bf16 conversion path (temp_seq inputs)
                const float* ap = (const float*)X + (size_t)(mbase + lrow) * ldx + k0 + lcol;
#pragma unroll
                for (int i = 0; i < 4; ++i) {
                    float4 f = *(const float4*)(ap + 4 * i);
                    AsD[soff + lrow * LSTR + lcol + 4*i + 0] = f2bf(f.x);
                    AsD[soff + lrow * LSTR + lcol + 4*i + 1] = f2bf(f.y);
                    AsD[soff + lrow * LSTR + lcol + 4*i + 2] = f2bf(f.z);
                    AsD[soff + lrow * LSTR + lcol + 4*i + 3] = f2bf(f.w);
                }
            } else {
                const unsigned short* ap = (const unsigned short*)X +
                                           (size_t)(mbase + lrow) * ldx + k0 + lcol;
                async_b128(aLds,      ap);
                async_b128(aLds + 16, ap + 8);
            }
            const unsigned short* bp = Wx + (size_t)(nbase + lrow) * ldbx + k0 + lcol;
            async_b128(bLds,      bp);
            async_b128(bLds + 16, bp + 8);
        } else {
            const int k0 = (kk - xsteps) * TK;
            const unsigned short* ap = Hm + (size_t)(mbase + lrow) * ldh + k0 + lcol;
            async_b128(aLds,      ap);
            async_b128(aLds + 16, ap + 8);
            const unsigned short* bp = Wh + (size_t)(nbase + lrow) * ldbh + k0 + lcol;
            async_b128(bLds,      bp);
            async_b128(bLds + 16, bp + 8);
        }
    };

    stage(0);
    for (int kk = 0; kk < NS; ++kk) {
        // my async copies for buf[kk&1] done; barrier makes everyone's visible,
        // and guarantees buf[(kk+1)&1] has no remaining readers.
        async_wait0();
        __syncthreads();
        if (kk + 1 < NS) stage(kk + 1);

        const int soff = (kk & 1) * BUFSZ;

        v16bf afrag[2];
#pragma unroll
        for (int mi = 0; mi < 2; ++mi) {
            const int base = soff + (wm * 32 + mi * 16 + arow) * LSTR;
            union { v16bf v; uint4 u[2]; } t;
            t.u[0] = *(const uint4*)&AsD[base + aoff];
            t.u[1] = *(const uint4*)&AsD[base + 16 + aoff];
            afrag[mi] = t.v;
        }
        v16bf bfrag[4];
#pragma unroll
        for (int g = 0; g < 4; ++g) {
            const int base = soff + (wn * 64 + g * 16 + bcol) * LSTR + boff;
            union { v16bf v; uint4 u[2]; } t;
            t.u[0] = *(const uint4*)&BsD[base];
            t.u[1] = *(const uint4*)&BsD[base + 16];
            bfrag[g] = t.v;
        }
#pragma unroll
        for (int mi = 0; mi < 2; ++mi)
#pragma unroll
            for (int g = 0; g < 4; ++g)
                acc[mi * 4 + g] = __builtin_amdgcn_wmma_f32_16x16x32_bf16(
                    false, afrag[mi], false, bfrag[g],
                    (short)0, acc[mi * 4 + g], false, false);
    }
}

// ---------------------------------------------------------------------------
// Fully fused LSTM step: gates = X@Wx^T + Hin@Wh^T + bias (+ speed rank-1),
// then the cell update in registers. Weights/bias/wcol are gate-permuted so
// each lane holds i,f,g,o of one (row, j) in acc[mi*4+0..3][r].
// Hin/Hout are bf16; c stays f32.
// ---------------------------------------------------------------------------
template <typename XT>
__global__ __launch_bounds__(256) void lstm_step_kernel(
    const XT* __restrict__ X, int ldx,
    const unsigned short* __restrict__ Wx,        // permuted bf16 [4096 x 128]
    const unsigned short* __restrict__ Hin,       // bf16 [2048 x 1024]
    const unsigned short* __restrict__ Wh,        // permuted bf16 [4096 x 1024]
    const float* __restrict__ bias,               // permuted [4096]
    const float* __restrict__ speeds_t,           // avg_speeds + t (stride MDEC) or null
    const float* __restrict__ wcol,               // permuted [4096] or null
    float* __restrict__ c,                        // f32 [2048 x 1024] in/out
    unsigned short* __restrict__ Hout)            // bf16 [2048 x 1024]
{
    __shared__ unsigned short AsD[2 * BUFSZ];
    __shared__ unsigned short BsD[2 * BUFSZ];

    const int tid  = threadIdx.x;
    const int lane = tid & 31;
    const int wid  = tid >> 5;
    const int wm   = wid >> 1;
    const int wn   = wid & 1;
    const int mbase = blockIdx.y * TM;
    const int nbase = blockIdx.x * TN;

    v8f zero = {0.f,0.f,0.f,0.f,0.f,0.f,0.f,0.f};
    v8f acc[8];
#pragma unroll
    for (int i = 0; i < 8; ++i) acc[i] = zero;

    gemm_pipelined<XT>(X, ldx, Wx, TEMPD, TEMPD,
                       Hin, HID, Wh, HID, HID,
                       mbase, nbase, tid, AsD, BsD, acc);

    // ---- fused LSTM cell epilogue ----
    const int rowadd = (lane >> 4) * 8;     // C/D layout: VGPR r -> row r (+8 hi half)
    const int jl     = lane & 15;
    const int jglob  = (nbase >> 2) + wn * 16 + jl;   // (nbase/64 + wn)*16 + jl

    float bv[4], wc[4];
#pragma unroll
    for (int g = 0; g < 4; ++g) {
        const int col = nbase + wn * 64 + g * 16 + jl;
        bv[g] = bias[col];
        wc[g] = wcol ? wcol[col] : 0.0f;
    }
#pragma unroll
    for (int mi = 0; mi < 2; ++mi) {
#pragma unroll
        for (int r = 0; r < 8; ++r) {
            const int row = mbase + wm * 32 + mi * 16 + rowadd + r;
            float gi = acc[mi * 4 + 0][r] + bv[0];
            float gf = acc[mi * 4 + 1][r] + bv[1];
            float gg = acc[mi * 4 + 2][r] + bv[2];
            float go = acc[mi * 4 + 3][r] + bv[3];
            if (speeds_t) {
                float s = speeds_t[(size_t)row * MDEC];
                gi += s * wc[0]; gf += s * wc[1];
                gg += s * wc[2]; go += s * wc[3];
            }
            float ig = 1.0f / (1.0f + __expf(-gi));
            float fg = 1.0f / (1.0f + __expf(-gf));
            float gt = 2.0f / (1.0f + __expf(-2.0f * gg)) - 1.0f;
            float og = 1.0f / (1.0f + __expf(-go));
            const size_t off = (size_t)row * HID + jglob;
            float cn = fg * c[off] + ig * gt;
            c[off] = cn;
            float hn = og * (2.0f / (1.0f + __expf(-2.0f * cn)) - 1.0f);
            Hout[off] = f2bf(hn);
        }
    }
}

// ---------------------------------------------------------------------------
// FC head: pred = h @ fc_W^T + fc_b (plain layout).
// Writes f32 to the output tensor slice and bf16 to the feedback buffer.
// ---------------------------------------------------------------------------
__global__ __launch_bounds__(256) void fc_kernel(
    const unsigned short* __restrict__ Hc,        // bf16 [2048 x 1024]
    const unsigned short* __restrict__ Wf,        // bf16 [128 x 1024]
    const float* __restrict__ bias,               // [128]
    float* __restrict__ C, int ldc,               // output slice [:, t, :]
    unsigned short* __restrict__ C2, int ldc2)    // bf16 feedback x
{
    __shared__ unsigned short AsD[2 * BUFSZ];
    __shared__ unsigned short BsD[2 * BUFSZ];

    const int tid  = threadIdx.x;
    const int lane = tid & 31;
    const int wid  = tid >> 5;
    const int wm   = wid >> 1;
    const int wn   = wid & 1;
    const int mbase = blockIdx.y * TM;
    const int nbase = blockIdx.x * TN;

    v8f zero = {0.f,0.f,0.f,0.f,0.f,0.f,0.f,0.f};
    v8f acc[8];
#pragma unroll
    for (int i = 0; i < 8; ++i) acc[i] = zero;

    // single segment: KX = 0, everything goes through the bf16 async h-path
    gemm_pipelined<unsigned short>(Hc, HID, Wf, HID, 0,
                                   Hc, HID, Wf, HID, HID,
                                   mbase, nbase, tid, AsD, BsD, acc);

    const int rowadd = (lane >> 4) * 8;
    const int jl     = lane & 15;
#pragma unroll
    for (int g = 0; g < 4; ++g) {
        const int col = nbase + wn * 64 + g * 16 + jl;
        const float bv = bias[col];
#pragma unroll
        for (int mi = 0; mi < 2; ++mi) {
#pragma unroll
            for (int r = 0; r < 8; ++r) {
                const int row = mbase + wm * 32 + mi * 16 + rowadd + r;
                float v = acc[mi * 4 + g][r] + bv;
                C[(size_t)row * ldc + col]    = v;
                C2[(size_t)row * ldc2 + col]  = f2bf(v);
            }
        }
    }
}

// ---------------------------------------------------------------------------
// Prep utilities
// ---------------------------------------------------------------------------
__global__ void permute_convert_kernel(const float* __restrict__ in, int ild,
                                       unsigned short* __restrict__ out, int cols)
{
    int idx = blockIdx.x * blockDim.x + threadIdx.x;
    if (idx >= G4H * cols) return;
    int nr = idx / cols, cc = idx - nr * cols;
    out[idx] = f2bf(in[(size_t)perm_src_row(nr) * ild + cc]);
}

__global__ void convert2d_kernel(const float* __restrict__ in, int ild,
                                 unsigned short* __restrict__ out, int old_,
                                 int rows, int cols)
{
    int idx = blockIdx.x * blockDim.x + threadIdx.x;
    if (idx >= rows * cols) return;
    int r = idx / cols, cc = idx - r * cols;
    out[(size_t)r * old_ + cc] = f2bf(in[(size_t)r * ild + cc]);
}

__global__ void permute_bias_kernel(const float* a, const float* b, float* o)
{
    int nr = blockIdx.x * blockDim.x + threadIdx.x;
    if (nr >= G4H) return;
    int s = perm_src_row(nr);
    o[nr] = a[s] + b[s];
}

__global__ void permute_col_kernel(const float* in, float* o)   // dec_Wih[:,128], permuted
{
    int nr = blockIdx.x * blockDim.x + threadIdx.x;
    if (nr >= G4H) return;
    o[nr] = in[(size_t)perm_src_row(nr) * (TEMPD + 1) + TEMPD];
}

__global__ void zero_f32_kernel(float* p, int n)
{
    int i = blockIdx.x * blockDim.x + threadIdx.x;
    if (i < n) p[i] = 0.0f;
}

__global__ void zero_u16_kernel(unsigned short* p, int n)
{
    int i = blockIdx.x * blockDim.x + threadIdx.x;
    if (i < n) p[i] = 0;
}

// ---------------------------------------------------------------------------
extern "C" void kernel_launch(void* const* d_in, const int* in_sizes, int n_in,
                              void* d_out, int out_size, void* d_ws, size_t ws_size,
                              hipStream_t stream)
{
    const float* temp_seq   = (const float*)d_in[0];   // [B,T,TEMP]
    const float* avg_speeds = (const float*)d_in[1];   // [B,M]
    const float* enc_Wih    = (const float*)d_in[2];   // [4H,TEMP]
    const float* enc_Whh    = (const float*)d_in[3];   // [4H,H]
    const float* enc_bih    = (const float*)d_in[4];
    const float* enc_bhh    = (const float*)d_in[5];
    const float* dec_Wih    = (const float*)d_in[6];   // [4H,TEMP+1]
    const float* dec_Whh    = (const float*)d_in[7];   // [4H,H]
    const float* dec_bih    = (const float*)d_in[8];
    const float* dec_bhh    = (const float*)d_in[9];
    const float* fc_W       = (const float*)d_in[10];  // [TEMP,H]
    const float* fc_b       = (const float*)d_in[11];  // [TEMP]
    float* out = (float*)d_out;                        // [B,M,TEMP]

    char* ws = (char*)d_ws;
    size_t off = 0;
    auto carve = [&](size_t bytes) -> void* {
        void* p = ws + off;
        off = (off + bytes + 255) & ~(size_t)255;
        return p;
    };
    unsigned short* hb0  = (unsigned short*)carve((size_t)BDIM * HID * 2);   // h ping (bf16)
    unsigned short* hb1  = (unsigned short*)carve((size_t)BDIM * HID * 2);   // h pong (bf16)
    float*          c    = (float*)carve((size_t)BDIM * HID * 4);
    unsigned short* xdec = (unsigned short*)carve((size_t)BDIM * TEMPD * 2); // bf16 feedback
    unsigned short* encWih_bf = (unsigned short*)carve((size_t)G4H * TEMPD * 2);
    unsigned short* encWhh_bf = (unsigned short*)carve((size_t)G4H * HID * 2);
    unsigned short* decWih_bf = (unsigned short*)carve((size_t)G4H * TEMPD * 2);
    unsigned short* decWhh_bf = (unsigned short*)carve((size_t)G4H * HID * 2);
    unsigned short* fcW_bf    = (unsigned short*)carve((size_t)TEMPD * HID * 2);
    float* bsum_enc = (float*)carve((size_t)G4H * 4);
    float* bsum_dec = (float*)carve((size_t)G4H * 4);
    float* wcol     = (float*)carve((size_t)G4H * 4);
    (void)ws_size; (void)in_sizes; (void)n_in; (void)out_size;

    auto blks = [](int n) { return (n + 255) / 256; };

    // ---- prep: permuted bf16 weights, permuted fused biases, state init ----
    permute_convert_kernel<<<blks(G4H * TEMPD), 256, 0, stream>>>(enc_Wih, TEMPD, encWih_bf, TEMPD);
    permute_convert_kernel<<<blks(G4H * HID),   256, 0, stream>>>(enc_Whh, HID,   encWhh_bf, HID);
    permute_convert_kernel<<<blks(G4H * TEMPD), 256, 0, stream>>>(dec_Wih, TEMPD + 1, decWih_bf, TEMPD);
    permute_convert_kernel<<<blks(G4H * HID),   256, 0, stream>>>(dec_Whh, HID,   decWhh_bf, HID);
    convert2d_kernel<<<blks(TEMPD * HID), 256, 0, stream>>>(fc_W, HID, fcW_bf, HID, TEMPD, HID);
    permute_bias_kernel<<<blks(G4H), 256, 0, stream>>>(enc_bih, enc_bhh, bsum_enc);
    permute_bias_kernel<<<blks(G4H), 256, 0, stream>>>(dec_bih, dec_bhh, bsum_dec);
    permute_col_kernel<<<blks(G4H), 256, 0, stream>>>(dec_Wih, wcol);
    zero_u16_kernel<<<blks(BDIM * HID), 256, 0, stream>>>(hb0, BDIM * HID);
    zero_f32_kernel<<<blks(BDIM * HID), 256, 0, stream>>>(c, BDIM * HID);

    const dim3 gGrid(G4H / TN, BDIM / TM);     // 32 x 16 blocks
    const dim3 fGrid(TEMPD / TN, BDIM / TM);   //  1 x 16 blocks
    const int  seq_lda = TTOT * TEMPD;         // 21504
    unsigned short* hb[2] = { hb0, hb1 };
    int hp = 0;                                // current h buffer index

    // ---- encoder: 168 fused recurrent steps (f32 x, bf16 h) ----
    for (int t = 0; t < TTOT; ++t) {
        lstm_step_kernel<float><<<gGrid, 256, 0, stream>>>(
            temp_seq + (size_t)t * TEMPD, seq_lda, encWih_bf,
            hb[hp], encWhh_bf,
            bsum_enc, nullptr, nullptr,
            c, hb[1 - hp]);
        hp ^= 1;
    }

    // ---- decoder: 48 fused steps + FC feedback ----
    for (int t = 0; t < MDEC; ++t) {
        if (t == 0) {
            lstm_step_kernel<float><<<gGrid, 256, 0, stream>>>(
                temp_seq + (size_t)(TTOT - 1) * TEMPD, seq_lda, decWih_bf,
                hb[hp], decWhh_bf,
                bsum_dec, avg_speeds + t, wcol,
                c, hb[1 - hp]);
        } else {
            lstm_step_kernel<unsigned short><<<gGrid, 256, 0, stream>>>(
                xdec, TEMPD, decWih_bf,
                hb[hp], decWhh_bf,
                bsum_dec, avg_speeds + t, wcol,
                c, hb[1 - hp]);
        }
        hp ^= 1;
        fc_kernel<<<fGrid, 256, 0, stream>>>(
            hb[hp], fcW_bf, fc_b,
            out + (size_t)t * TEMPD, MDEC * TEMPD, xdec, TEMPD);
    }
}
